// MinRNN_2851858285199
// MI455X (gfx1250) — matrix operand order repaired and checked by reading, not compiled, not run on hardware
//
#include <hip/hip_runtime.h>

typedef __attribute__((ext_vector_type(16))) _Float16 v16h;
typedef __attribute__((ext_vector_type(8)))  _Float16 v8h;
typedef __attribute__((ext_vector_type(8)))  float    v8f;

// ---------------------------------------------------------------------------
// f32 -> f16 conversion
// ---------------------------------------------------------------------------
__global__ void f32_to_f16_kernel(const float* __restrict__ in,
                                  _Float16* __restrict__ out, size_t n) {
  size_t i = (size_t)blockIdx.x * blockDim.x + threadIdx.x;
  if (i < n) out[i] = (_Float16)in[i];
}

// ---------------------------------------------------------------------------
// B fragment loader (32x16 f16: halves [hi*8,+8) and [hi*8+16,+8) per lane).
// ---------------------------------------------------------------------------
__device__ __forceinline__ v16h load_bfrag(const _Float16* p) {
  v8h lo = *(const v8h*)(p);
  v8h h8 = *(const v8h*)(p + 16);
  v16h f;
#pragma unroll
  for (int e = 0; e < 8; ++e) { f[e] = lo[e]; f[e + 8] = h8[e]; }
  return f;
}

// LDS A-tile geometry: 128 rows x 32 halves, rows padded to 40 halves (80 B)
// to spread DS bank groups; two buffers for double-buffering.
#define LDS_ROWH 40
#define LDS_BUFH (128 * LDS_ROWH)

// Light workgroup barrier: waits only DScnt (LDS visibility), NOT loadcnt,
// so global prefetch loads stay in flight across the barrier.  Mnemonics
// match compiler-emitted gfx1250 code (s_barrier_signal/-wait -1).
#define LITE_SYNC()                                                     \
  asm volatile("s_wait_dscnt 0x0\n\t"                                   \
               "s_barrier_signal -1\n\t"                                \
               "s_barrier_wait -1" ::: "memory")

#define MMA_FRAGS(AF, BF)                                               \
  do {                                                                  \
    _Pragma("unroll") for (int i = 0; i < 4; ++i)                       \
      _Pragma("unroll") for (int j = 0; j < 2; ++j)                     \
        acc[i][j] = __builtin_amdgcn_wmma_f32_16x16x32_f16(             \
            false, AF[i], false, BF[j], (short)0, acc[i][j],            \
            false, false);                                              \
  } while (0)

// ---------------------------------------------------------------------------
// C[M,N] = act( A[M,K] @ W[N,K]^T + bias )
// Block: 256 threads = 8 waves (2 x 4); block tile 128 x 128.
// 3-stage pipeline, unrolled x2 (register sets alternate by parity):
//   step s: global-load tile s+2 -> regs | ds_store tile s+1 -> other buffer
//           | ds_load fragments of tile s | 8x WMMA | light barrier.
// Each wave computes 64x32 output via 4x2 f32 WMMA accumulators.
// K must be a multiple of 64.
// ---------------------------------------------------------------------------
__global__ __launch_bounds__(256, 1)
void wmma_gemm_nt(const _Float16* __restrict__ A,
                  const _Float16* __restrict__ W,
                  const float* __restrict__ bias,   // may be null, len N
                  float* __restrict__ Cf,           // may be null
                  _Float16* __restrict__ Ch,        // may be null
                  int M, int N, int K,
                  int ldc, int act /*0=none,1=tanh*/) {
  __shared__ __align__(16) _Float16 smemA[2 * LDS_BUFH];

  const int tid  = threadIdx.x;
  const int lane = tid & 31;
  const int wave = tid >> 5;
  const int wm = wave >> 2;                 // 0..1
  const int wn = wave & 3;                  // 0..3
  const int rowBase = blockIdx.y * 128;     // block's A rows
  const int bm = rowBase + wm * 64;
  const int bn = blockIdx.x * 128 + wn * 32;

  const int r  = lane & 15;
  const int hi = lane >> 4;

  const _Float16* wrow[2];
#pragma unroll
  for (int j = 0; j < 2; ++j)
    wrow[j] = W + (size_t)(bn + j * 16 + r) * K + hi * 8;

  // Cooperative A-tile staging: 512 x 16B chunks, 2 per thread.
  const int chunk0 = tid * 2, chunk1 = tid * 2 + 1;
  const int srow0 = chunk0 >> 2, spart0 = chunk0 & 3;   // row 0..127, part 0..3
  const int srow1 = chunk1 >> 2, spart1 = chunk1 & 3;
  const _Float16* sA0 = A + (size_t)(rowBase + srow0) * K + spart0 * 8;
  const _Float16* sA1 = A + (size_t)(rowBase + srow1) * K + spart1 * 8;
  const int dOff0 = srow0 * LDS_ROWH + spart0 * 8;      // halves
  const int dOff1 = srow1 * LDS_ROWH + spart1 * 8;

  // A fragments from LDS (ISA A 16x32 layout: elems 0..7 = halves [hi*8,+8),
  // elems 8..15 = halves [hi*8+16,+8) of the lane's row).
  auto ds_frags = [&](int bufBase, v16h af[4]) {
#pragma unroll
    for (int i = 0; i < 4; ++i) {
      int row = wm * 64 + i * 16 + r;
      const _Float16* p = smemA + bufBase + row * LDS_ROWH + hi * 8;
      v8h lo = *(const v8h*)(p);
      v8h h8 = *(const v8h*)(p + 16);
#pragma unroll
      for (int e = 0; e < 8; ++e) { af[i][e] = lo[e]; af[i][e + 8] = h8[e]; }
    }
  };

  v8f acc[4][2] = {};
  v16h bA[2], bB[2];
  v8h ga0, ga1, gb0, gb1;
  const int KS = K / 32;    // even, >= 2

  // Prologue: tile 0 -> LDS buf0; tile 1 -> registers; B frags for steps 0,1.
  ga0 = *(const v8h*)(sA0);
  ga1 = *(const v8h*)(sA1);
  *(v8h*)(smemA + dOff0) = ga0;
  *(v8h*)(smemA + dOff1) = ga1;
  ga0 = *(const v8h*)(sA0 + 32);
  ga1 = *(const v8h*)(sA1 + 32);
  bA[0] = load_bfrag(wrow[0]);
  bA[1] = load_bfrag(wrow[1]);
  bB[0] = load_bfrag(wrow[0] + 32);
  bB[1] = load_bfrag(wrow[1] + 32);
  LITE_SYNC();

  for (int s = 0; s < KS; s += 2) {
    // ---- even step: consume buf0 (tile s) with bA ----
    *(v8h*)(smemA + LDS_BUFH + dOff0) = ga0;   // buf1 <- tile s+1
    *(v8h*)(smemA + LDS_BUFH + dOff1) = ga1;
    if (s + 2 < KS) {
      gb0 = *(const v8h*)(sA0 + (s + 2) * 32); // prefetch tile s+2
      gb1 = *(const v8h*)(sA1 + (s + 2) * 32);
    }
    v16h af[4];
    ds_frags(0, af);
    MMA_FRAGS(af, bA);
    if (s + 2 < KS) {
      bA[0] = load_bfrag(wrow[0] + (s + 2) * 32);
      bA[1] = load_bfrag(wrow[1] + (s + 2) * 32);
    }
    LITE_SYNC();

    // ---- odd step: consume buf1 (tile s+1) with bB ----
    if (s + 2 < KS) {
      *(v8h*)(smemA + dOff0) = gb0;            // buf0 <- tile s+2
      *(v8h*)(smemA + dOff1) = gb1;
    }
    if (s + 3 < KS) {
      ga0 = *(const v8h*)(sA0 + (s + 3) * 32); // prefetch tile s+3
      ga1 = *(const v8h*)(sA1 + (s + 3) * 32);
    }
    v16h ag[4];
    ds_frags(LDS_BUFH, ag);
    MMA_FRAGS(ag, bB);
    if (s + 3 < KS) {
      bB[0] = load_bfrag(wrow[0] + (s + 3) * 32);
      bB[1] = load_bfrag(wrow[1] + (s + 3) * 32);
    }
    LITE_SYNC();
  }

  // Epilogue. 16x16 f32 C layout: lanes 0-15 N=lane, M=vr; lanes 16-31 M=vr+8.
#pragma unroll
  for (int i = 0; i < 4; ++i) {
#pragma unroll
    for (int j = 0; j < 2; ++j) {
      int col = bn + j * 16 + r;
      float bv = bias ? bias[col] : 0.f;
#pragma unroll
      for (int vr = 0; vr < 8; ++vr) {
        int row = bm + i * 16 + hi * 8 + vr;
        float v = acc[i][j][vr] + bv;
        if (act == 1) v = tanhf(v);
        size_t idx = (size_t)row * ldc + col;
        if (Cf) Cf[idx] = v;
        if (Ch) Ch[idx] = (_Float16)v;
      }
    }
  }
}

// ---------------------------------------------------------------------------
// Chunked diagonal scan, 3 phases.  Chunk length CH=64, C=T/CH chunks.
// mode 1 (iteration 1): h_prev == 0, u = sigmoid(ZU+bu)
// mode 2 (iteration 2): h_prev = Hprev (pre-shifted), u = sigmoid(G+ZU+bu)
// Composite per step from chunk start:  h_t = CA[t]*h_chunk0 + CB[t].
// ---------------------------------------------------------------------------
__device__ __forceinline__ void deer_step(float z, float zu, float g,
                                          float hp, float bub, float dw,
                                          bool first, float& As, float& bs) {
  float u = 1.f / (1.f + __expf(-(g + zu + bub)));
  float fs = u * hp + (1.f - u) * z;
  As = first ? 0.f : (u + (hp - z) * u * (1.f - u) * dw);
  bs = fs - As * hp;
}

__global__ __launch_bounds__(256)
void scan_phaseA(int mode, const _Float16* __restrict__ Z,
                 const float* __restrict__ ZU, const float* __restrict__ G,
                 const float* __restrict__ bu, const float* __restrict__ Wr,
                 const _Float16* __restrict__ Hprev,
                 float* __restrict__ CA, float* __restrict__ CB,
                 int Bc, int Tlen, int D, int CH, int dir) {
  int id = blockIdx.x * blockDim.x + threadIdx.x;
  int C = Tlen / CH;
  if (id >= Bc * C * D) return;
  int ch = id % D;
  int bc = id / D;
  int c = bc % C, b = bc / C;
  float dw = Wr[(size_t)ch * D + ch];
  float bub = bu[ch];
  float Aacc = 1.f, Bacc = 0.f;
  for (int sl = 0; sl < CH; ++sl) {
    int sg = c * CH + sl;
    int t = dir ? (Tlen - 1 - sg) : sg;
    size_t idx = ((size_t)b * Tlen + t) * D + ch;
    float z = (float)Z[idx];
    float g = (mode == 2) ? G[idx] : 0.f;
    float hp = (mode == 2) ? (float)Hprev[idx] : 0.f;
    float As, bs;
    deer_step(z, ZU[idx], g, hp, bub, dw, sg == 0, As, bs);
    Aacc = As * Aacc;
    Bacc = As * Bacc + bs;
    CA[idx] = Aacc;
    CB[idx] = Bacc;
  }
}

__global__ __launch_bounds__(256)
void scan_phaseB(const float* __restrict__ CA, const float* __restrict__ CB,
                 float* __restrict__ Hstart,
                 int Bc, int Tlen, int D, int CH, int dir) {
  int id = blockIdx.x * blockDim.x + threadIdx.x;
  if (id >= Bc * D) return;
  int ch = id % D, b = id / D;
  int C = Tlen / CH;
  float h = 0.f;
  for (int c = 0; c < C; ++c) {
    Hstart[((size_t)b * C + c) * D + ch] = h;
    int sg = c * CH + CH - 1;
    int t = dir ? (Tlen - 1 - sg) : sg;
    size_t idx = ((size_t)b * Tlen + t) * D + ch;
    h = CA[idx] * h + CB[idx];
  }
}

__global__ __launch_bounds__(256)
void scan_phaseC(int mode, const float* __restrict__ CA,
                 const float* __restrict__ CB, const float* __restrict__ Hstart,
                 _Float16* __restrict__ HprevOut,  // mode 1: pre-shifted states
                 _Float16* __restrict__ outH,      // mode 2 (may be null)
                 float* __restrict__ outF,         // mode 2 (may be null)
                 int Bc, int Tlen, int D, int CH, int dir, int ldo, int ocol) {
  int id = blockIdx.x * blockDim.x + threadIdx.x;
  int C = Tlen / CH;
  if (id >= Bc * C * D) return;
  int ch = id % D;
  int bc = id / D;
  int c = bc % C, b = bc / C;
  float h0 = Hstart[((size_t)b * C + c) * D + ch];
  if (mode == 1 && c == 0) {   // boundary of the pre-shifted buffer
    int tb = dir ? (Tlen - 1) : 0;
    HprevOut[((size_t)b * Tlen + tb) * D + ch] = (_Float16)0.f;
  }
  for (int sl = 0; sl < CH; ++sl) {
    int sg = c * CH + sl;
    int t = dir ? (Tlen - 1 - sg) : sg;
    size_t idx = ((size_t)b * Tlen + t) * D + ch;
    float h = CA[idx] * h0 + CB[idx];
    if (mode == 1) {
      int tw = dir ? (t - 1) : (t + 1);
      if (tw >= 0 && tw < Tlen)
        HprevOut[((size_t)b * Tlen + tw) * D + ch] = (_Float16)h;
    } else {
      size_t oidx = ((size_t)b * Tlen + t) * ldo + ocol + ch;
      if (outH) outH[oidx] = (_Float16)h;
      if (outF) outF[oidx] = h;
    }
  }
}

// ---------------------------------------------------------------------------
// h_n extraction: (4, B, D) appended after the main output.
// ---------------------------------------------------------------------------
__global__ void write_hn(const _Float16* __restrict__ L0out,  // (B*T, 2D) f16
                         const float* __restrict__ L1out,     // (B*T, 2D) f32
                         float* __restrict__ hn, int Bc, int Tlen, int D) {
  int id = blockIdx.x * blockDim.x + threadIdx.x;
  if (id >= Bc * D) return;
  int b = id / D, ch = id % D;
  size_t lastRow  = (size_t)b * Tlen + (Tlen - 1);
  size_t firstRow = (size_t)b * Tlen;
  int n = Bc * D;
  hn[0 * n + id] = (float)L0out[lastRow  * 2 * D + ch];
  hn[1 * n + id] = (float)L0out[firstRow * 2 * D + D + ch];
  hn[2 * n + id] = L1out[lastRow  * 2 * D + ch];
  hn[3 * n + id] = L1out[firstRow * 2 * D + D + ch];
}

// ---------------------------------------------------------------------------
extern "C" void kernel_launch(void* const* d_in, const int* in_sizes, int n_in,
                              void* d_out, int out_size, void* d_ws, size_t ws_size,
                              hipStream_t stream) {
  (void)in_sizes; (void)n_in; (void)out_size; (void)ws_size;
  const int B = 16, T = 1024, D = 512, I = 512, CH = 64;
  const int M = B * T;
  const int C = T / CH;

  const float* x   = (const float*)d_in[0];
  const float* Wi0 = (const float*)d_in[1];
  const float* bi0 = (const float*)d_in[2];
  const float* Wr0 = (const float*)d_in[3];
  const float* Uz0 = (const float*)d_in[4];
  const float* bu0 = (const float*)d_in[5];
  const float* Wi1 = (const float*)d_in[6];
  const float* bi1 = (const float*)d_in[7];
  const float* Wr1 = (const float*)d_in[8];
  const float* Uz1 = (const float*)d_in[9];
  const float* bu1 = (const float*)d_in[10];

  // Workspace carve-up (256B aligned), ~215.5 MiB total.
  char* ws = (char*)d_ws;
  size_t off = 0;
  auto alloc = [&](size_t bytes) -> void* {
    void* p = ws + off;
    off += (bytes + 255) & ~(size_t)255;
    return p;
  };
  _Float16* hWi0  = (_Float16*)alloc((size_t)2 * D * I * 2);
  _Float16* hWr0  = (_Float16*)alloc((size_t)2 * D * D * 2);
  _Float16* hUz0  = (_Float16*)alloc((size_t)2 * D * D * 2);
  _Float16* hWi1  = (_Float16*)alloc((size_t)2 * D * 2 * D * 2);
  _Float16* hWr1  = (_Float16*)alloc((size_t)2 * D * D * 2);
  _Float16* hUz1  = (_Float16*)alloc((size_t)2 * D * D * 2);
  _Float16* hX0   = (_Float16*)alloc((size_t)M * I * 2);
  _Float16* hX1   = (_Float16*)alloc((size_t)M * 2 * D * 2);
  _Float16* hZ    = (_Float16*)alloc((size_t)M * D * 2);
  float*    ZU    = (float*)   alloc((size_t)M * D * 4);
  float*    G     = (float*)   alloc((size_t)M * D * 4);
  _Float16* Hprev = (_Float16*)alloc((size_t)M * D * 2);
  float*    CAb   = (float*)   alloc((size_t)M * D * 4);
  float*    CBb   = (float*)   alloc((size_t)M * D * 4);
  float*    Hstart= (float*)   alloc((size_t)B * C * D * 4);

  auto cvt = [&](const float* src, _Float16* dst, size_t n) {
    f32_to_f16_kernel<<<(unsigned)((n + 255) / 256), 256, 0, stream>>>(src, dst, n);
  };
  cvt(x,   hX0,  (size_t)M * I);
  cvt(Wi0, hWi0, (size_t)2 * D * I);
  cvt(Wr0, hWr0, (size_t)2 * D * D);
  cvt(Uz0, hUz0, (size_t)2 * D * D);
  cvt(Wi1, hWi1, (size_t)2 * D * 2 * D);
  cvt(Wr1, hWr1, (size_t)2 * D * D);
  cvt(Uz1, hUz1, (size_t)2 * D * D);

  dim3 ggrid(D / 128, M / 128);                  // (4, 128)
  int blkBD  = (B * D + 255) / 256;              // 8192 threads
  int blkBCD = (B * C * D + 255) / 256;          // 131072 threads

  for (int l = 0; l < 2; ++l) {
    int Din = (l == 0) ? I : 2 * D;
    const _Float16* Xh  = (l == 0) ? hX0 : hX1;
    const _Float16* Wih = (l == 0) ? hWi0 : hWi1;
    const _Float16* Wrh = (l == 0) ? hWr0 : hWr1;
    const _Float16* Uzh = (l == 0) ? hUz0 : hUz1;
    const float* biF = (l == 0) ? bi0 : bi1;
    const float* buF = (l == 0) ? bu0 : bu1;
    const float* WrF = (l == 0) ? Wr0 : Wr1;

    for (int dr = 0; dr < 2; ++dr) {
      const _Float16* Wid = Wih + (size_t)dr * D * Din;
      const _Float16* Uzd = Uzh + (size_t)dr * D * D;
      const _Float16* Wrd = Wrh + (size_t)dr * D * D;
      const float* bid = biF + (size_t)dr * D;
      const float* bud = buF + (size_t)dr * D;
      const float* Wrf = WrF + (size_t)dr * D * D;

      // 1) Z = tanh(X @ Wi^T + bi)  (f16 out)
      wmma_gemm_nt<<<ggrid, 256, 0, stream>>>(Xh, Wid, bid, nullptr, hZ,
                                              M, D, Din, D, 1);
      // 2) ZU = Z @ Uz^T  (f32 out)
      wmma_gemm_nt<<<ggrid, 256, 0, stream>>>(hZ, Uzd, nullptr, ZU, nullptr,
                                              M, D, D, D, 0);
      // 3) iteration 1: chunked scan -> Hprev (pre-shifted)
      scan_phaseA<<<blkBCD, 256, 0, stream>>>(1, hZ, ZU, nullptr, bud, Wrf,
                                              nullptr, CAb, CBb, B, T, D, CH, dr);
      scan_phaseB<<<blkBD, 256, 0, stream>>>(CAb, CBb, Hstart, B, T, D, CH, dr);
      scan_phaseC<<<blkBCD, 256, 0, stream>>>(1, CAb, CBb, Hstart, Hprev,
                                              nullptr, nullptr,
                                              B, T, D, CH, dr, 0, 0);
      // 4) G = Hprev @ Wr^T
      wmma_gemm_nt<<<ggrid, 256, 0, stream>>>(Hprev, Wrd, nullptr, G, nullptr,
                                              M, D, D, D, 0);
      // 5) iteration 2: chunked scan -> layer output
      _Float16* oh = (l == 0) ? hX1 : nullptr;
      float*    of = (l == 1) ? (float*)d_out : nullptr;
      scan_phaseA<<<blkBCD, 256, 0, stream>>>(2, hZ, ZU, G, bud, Wrf,
                                              Hprev, CAb, CBb, B, T, D, CH, dr);
      scan_phaseB<<<blkBD, 256, 0, stream>>>(CAb, CBb, Hstart, B, T, D, CH, dr);
      scan_phaseC<<<blkBCD, 256, 0, stream>>>(2, CAb, CBb, Hstart, nullptr,
                                              oh, of, B, T, D, CH, dr,
                                              2 * D, dr * D);
    }
  }

  write_hn<<<blkBD, 256, 0, stream>>>(hX1, (const float*)d_out,
                                      (float*)d_out + (size_t)M * 2 * D,
                                      B, T, D);
}
